// RejoinViG_28321014350042
// MI455X (gfx1250) — matrix operand order repaired and checked by reading, not compiled, not run on hardware
//
#include <hip/hip_runtime.h>
#include <math.h>

typedef __bf16 bf16;
typedef __attribute__((ext_vector_type(16))) __bf16 v16bf;
typedef __attribute__((ext_vector_type(8)))  __bf16 v8bf;
typedef __attribute__((ext_vector_type(8)))  float  v8f;

#define BB   8
#define CC   256
#define HH   28
#define WW2  28
#define HW   (HH * WW2)     // 784
#define PTOT (BB * HW)      // 6272
#define BN_EPS 1e-5f

// ---------------------------------------------------------------------------
// block-wide sum over 256 threads (8 wave32s)
// ---------------------------------------------------------------------------
static __device__ __forceinline__ float block_reduce_sum(float v) {
  __shared__ float s[8];
  #pragma unroll
  for (int o = 16; o > 0; o >>= 1) v += __shfl_xor(v, o, 32);
  int w = threadIdx.x >> 5, l = threadIdx.x & 31;
  __syncthreads();              // protect s[] across repeated calls
  if (l == 0) s[w] = v;
  __syncthreads();
  return s[0] + s[1] + s[2] + s[3] + s[4] + s[5] + s[6] + s[7];
}

// ---------------------------------------------------------------------------
// NCHW -> NHWC (pixel-major, channel-contiguous)
// ---------------------------------------------------------------------------
__global__ void k_transpose(const float* __restrict__ x, float* __restrict__ xn) {
  int p = blockIdx.x;           // 0..PTOT-1
  int c = threadIdx.x;          // 0..255
  int b = p / HW, hw = p % HW;
  xn[(size_t)p * CC + c] = x[((size_t)b * CC + c) * HW + hw];
}

__global__ void k_f32_to_bf16(const float* __restrict__ src, bf16* __restrict__ dst, int n) {
  int i = blockIdx.x * 256 + threadIdx.x;
  if (i < n) dst[i] = (bf16)src[i];
}

// ---------------------------------------------------------------------------
// CPE: depthwise 7x7 (zero pad 3) + bias + residual -> bf16 GEMM operand A1
// ---------------------------------------------------------------------------
__global__ void k_cpe(const float* __restrict__ xn, const float* __restrict__ w,
                      const float* __restrict__ bias, bf16* __restrict__ A1) {
  int p = blockIdx.x, c = threadIdx.x;
  int b = p / HW, hw = p % HW, h = hw / WW2, ww = hw % WW2;
  const float* xb = xn + (size_t)b * HW * CC;
  float acc = bias[c];
  #pragma unroll
  for (int dh = 0; dh < 7; dh++) {
    int hh = h + dh - 3;
    if (hh < 0 || hh >= HH) continue;
    #pragma unroll
    for (int dw = 0; dw < 7; dw++) {
      int wc = ww + dw - 3;
      if (wc < 0 || wc >= WW2) continue;
      acc += xb[((size_t)(hh * WW2 + wc)) * CC + c] * w[c * 49 + dh * 7 + dw];
    }
  }
  acc += xb[(size_t)hw * CC + c];
  A1[(size_t)p * CC + c] = (bf16)acc;
}

// ---------------------------------------------------------------------------
// per-batch threshold:  thr = mean - std(ddof=1) of ||x - roll(x,(14,14))||_1
// ---------------------------------------------------------------------------
__global__ void k_thr(const float* __restrict__ xn, float* __restrict__ thr) {
  int b = blockIdx.x;
  const float* xb = xn + (size_t)b * HW * CC;
  float sum = 0.f, sq = 0.f;
  for (int p = threadIdx.x; p < HW; p += 256) {
    int h = p / WW2, w = p % WW2;
    int sp = ((h + 14) % HH) * WW2 + (w + 14) % WW2;
    const float* a = xb + (size_t)p * CC;
    const float* r = xb + (size_t)sp * CC;
    float d = 0.f;
    for (int c = 0; c < CC; c++) d += fabsf(a[c] - r[c]);
    sum += d; sq += d * d;
  }
  sum = block_reduce_sum(sum);
  sq  = block_reduce_sum(sq);
  if (threadIdx.x == 0) {
    float mean = sum / (float)HW;
    float var  = (sq - sum * sum / (float)HW) / (float)(HW - 1);
    thr[b] = mean - sqrtf(fmaxf(var, 0.f));
  }
}

// ---------------------------------------------------------------------------
// MR aggregation: running max of masked (rolled - x); writes bf16 [x | xj]
// mode 0: local shifts (h:0,1,2  w:0,1,2)   mode 1: global (h,w: 0,2,...,26)
// ---------------------------------------------------------------------------
__global__ void k_mr(const float* __restrict__ xn, const float* __restrict__ thr,
                     bf16* __restrict__ A, int mode) {
  int p = blockIdx.x, c = threadIdx.x;
  int b = p / HW, hw = p % HW, h = hw / WW2, w = hw % WW2;
  const float* xb = xn + (size_t)b * HW * CC;
  float xc = xb[(size_t)hw * CC + c];
  float t = thr[b];
  float xj = 0.f;
  int nsh = mode ? 28 : 6;
  for (int k = 0; k < nsh; k++) {
    int axis, s;
    if (mode) { axis = k / 14; s = (k % 14) * 2; }
    else      { axis = k / 3;  s = k % 3; }
    int hh = axis ? h : (h - s + HH) % HH;
    int wc = axis ? (w - s + WW2) % WW2 : w;
    float r = xb[((size_t)(hh * WW2 + wc)) * CC + c];
    float dist = block_reduce_sum(fabsf(xc - r));
    if (dist < t) xj = fmaxf(xj, r - xc);
  }
  A[(size_t)p * 512 + c]       = (bf16)xc;
  A[(size_t)p * 512 + 256 + c] = (bf16)xj;
}

// ---------------------------------------------------------------------------
// NodeAttention: tanh channel-sum scores over 27 shifts, softmax, weighted sum
// writes bf16 [x | out]
// ---------------------------------------------------------------------------
__global__ void k_na(const float* __restrict__ xn, bf16* __restrict__ A) {
  int p = blockIdx.x, c = threadIdx.x;
  int b = p / HW, hw = p % HW, h = hw / WW2, w = hw % WW2;
  const float* xb = xn + (size_t)b * HW * CC;
  float xc = xb[(size_t)hw * CC + c];
  const float scale = 0.19245008973f;  // 27^-0.5
  __shared__ float sc[27];
  for (int k = 0; k < 27; k++) {
    int axis = (k >= 14);
    int s = axis ? (k - 13) * 2 : k * 2;
    int hh = axis ? h : (h - s + HH) % HH;
    int wc = axis ? (w - s + WW2) % WW2 : w;
    float r = xb[((size_t)(hh * WW2 + wc)) * CC + c];
    float sm = block_reduce_sum(tanhf(r * xc * scale));
    if (threadIdx.x == 0) sc[k] = sm;
  }
  __syncthreads();
  float mx = -1e30f;
  #pragma unroll
  for (int k = 0; k < 27; k++) mx = fmaxf(mx, sc[k]);
  float den = 0.f;
  #pragma unroll
  for (int k = 0; k < 27; k++) den += __expf(sc[k] - mx);
  float out = 0.f;
  for (int k = 0; k < 27; k++) {
    int axis = (k >= 14);
    int s = axis ? (k - 13) * 2 : k * 2;
    int hh = axis ? h : (h - s + HH) % HH;
    int wc = axis ? (w - s + WW2) % WW2 : w;
    float r = xb[((size_t)(hh * WW2 + wc)) * CC + c];
    out += r * (__expf(sc[k] - mx) / den);
  }
  A[(size_t)p * 512 + c]       = (bf16)xc;
  A[(size_t)p * 512 + 256 + c] = (bf16)out;
}

// ---------------------------------------------------------------------------
// WMMA GEMM with async-to-LDS weight staging.
// out[m][n] = act( (sum_k A[m][k]*W[n][k] + bias[n]) * g[n]/sqrt(1+eps) + be[n] )
// block = 256 threads = 8 waves (4M x 2N), wave tile 32x32, block tile 128x64.
// B tile (64 rows x K, contiguous in W) staged via GLOBAL_LOAD_ASYNC_TO_LDS.
// ---------------------------------------------------------------------------
__device__ __forceinline__ void epilogue(
    v8f acc, int mbase, int n, float sc, float bi, float be,
    float* __restrict__ out_f32, bf16* __restrict__ out_bf16,
    float* __restrict__ out_nchw, int do_gelu) {
  union { v8f v; float f[8]; } u;
  u.v = acc;
  #pragma unroll
  for (int v = 0; v < 8; v++) {
    int m = mbase + v;
    float val = (u.f[v] + bi) * sc + be;
    if (do_gelu) val = 0.5f * val * (1.f + erff(val * 0.70710678118f));
    if (out_f32)  out_f32[(size_t)m * 256 + n] = val;
    if (out_bf16) out_bf16[(size_t)m * 256 + n] = (bf16)val;
    if (out_nchw) {
      int b = m / HW, hw = m % HW;
      out_nchw[((size_t)b * 256 + n) * HW + hw] = val;
    }
  }
}

__global__ __launch_bounds__(256) void k_gemm(
    const bf16* __restrict__ A, const bf16* __restrict__ Wt,
    const float* __restrict__ bias, const float* __restrict__ gamma,
    const float* __restrict__ beta,
    float* __restrict__ out_f32, bf16* __restrict__ out_bf16,
    float* __restrict__ out_nchw, int K, int do_gelu) {
  __shared__ alignas(32) bf16 Bs[64 * 512];   // up to 64KB weight tile

  int bm = blockIdx.y * 128;
  int bn = blockIdx.x * 64;
  int tid = threadIdx.x;
  int wave = tid >> 5, lane = tid & 31;
  int wm = (wave >> 1) * 32, wn = (wave & 1) * 32;
  int lm = lane & 15, lhi = lane >> 4;

  // --- async-stage the 64 x K weight tile (rows bn..bn+63 are contiguous) ---
  {
    const bf16* wt = Wt + (size_t)bn * K;
    int total = 64 * K;
    for (int i = tid * 8; i < total; i += 256 * 8) {
      unsigned lds = (unsigned)(size_t)(void*)(Bs + i);
      const void* g = (const void*)(wt + i);
      asm volatile("global_load_async_to_lds_b128 %0, %1, off"
                   :: "v"(lds), "v"(g) : "memory");
    }
    asm volatile("s_wait_asynccnt 0" ::: "memory");
  }
  __syncthreads();

  v8f acc00 = {}, acc01 = {}, acc10 = {}, acc11 = {};
  const bf16* a0p = A + (size_t)(bm + wm + lm) * K;
  const bf16* a1p = A + (size_t)(bm + wm + 16 + lm) * K;
  const bf16* b0s = Bs + (size_t)(wn + lm) * K;
  const bf16* b1s = Bs + (size_t)(wn + 16 + lm) * K;

  for (int k0 = 0; k0 < K; k0 += 32) {
    __builtin_prefetch(a0p + k0 + 128, 0, 3);   // WGP-scope prefetch of next A
    __builtin_prefetch(a1p + k0 + 128, 0, 3);
    // A fragment: halves 0..7 -> K = k0+8*hi+0..7 ; halves 8..15 -> K = k0+16+8*hi+0..7
    v8bf a0lo = *(const v8bf*)(a0p + k0 + 8 * lhi);
    v8bf a0hi = *(const v8bf*)(a0p + k0 + 16 + 8 * lhi);
    v8bf a1lo = *(const v8bf*)(a1p + k0 + 8 * lhi);
    v8bf a1hi = *(const v8bf*)(a1p + k0 + 16 + 8 * lhi);
    // B fragment from LDS: halves 0..7 -> K = k0+8*hi+0..7 ; halves 8..15 -> K = k0+16+8*hi+0..7
    v8bf b0lo = *(const v8bf*)(b0s + k0 + 8 * lhi);
    v8bf b0hi = *(const v8bf*)(b0s + k0 + 16 + 8 * lhi);
    v8bf b1lo = *(const v8bf*)(b1s + k0 + 8 * lhi);
    v8bf b1hi = *(const v8bf*)(b1s + k0 + 16 + 8 * lhi);
    v16bf a0 = __builtin_shufflevector(a0lo, a0hi, 0,1,2,3,4,5,6,7,8,9,10,11,12,13,14,15);
    v16bf a1 = __builtin_shufflevector(a1lo, a1hi, 0,1,2,3,4,5,6,7,8,9,10,11,12,13,14,15);
    v16bf b0 = __builtin_shufflevector(b0lo, b0hi, 0,1,2,3,4,5,6,7,8,9,10,11,12,13,14,15);
    v16bf b1 = __builtin_shufflevector(b1lo, b1hi, 0,1,2,3,4,5,6,7,8,9,10,11,12,13,14,15);
    acc00 = __builtin_amdgcn_wmma_f32_16x16x32_bf16(false, a0, false, b0, (short)0, acc00, false, false);
    acc01 = __builtin_amdgcn_wmma_f32_16x16x32_bf16(false, a0, false, b1, (short)0, acc01, false, false);
    acc10 = __builtin_amdgcn_wmma_f32_16x16x32_bf16(false, a1, false, b0, (short)0, acc10, false, false);
    acc11 = __builtin_amdgcn_wmma_f32_16x16x32_bf16(false, a1, false, b1, (short)0, acc11, false, false);
  }

  const float rs = rsqrtf(1.f + BN_EPS);
  int n0 = bn + wn + lm;
  int n1 = bn + wn + 16 + lm;
  float s0 = gamma[n0] * rs, bi0 = bias[n0], be0 = beta[n0];
  float s1 = gamma[n1] * rs, bi1 = bias[n1], be1 = beta[n1];
  int m0 = bm + wm + lhi * 8;
  int m1 = bm + wm + 16 + lhi * 8;
  epilogue(acc00, m0, n0, s0, bi0, be0, out_f32, out_bf16, out_nchw, do_gelu);
  epilogue(acc01, m0, n1, s1, bi1, be1, out_f32, out_bf16, out_nchw, do_gelu);
  epilogue(acc10, m1, n0, s0, bi0, be0, out_f32, out_bf16, out_nchw, do_gelu);
  epilogue(acc11, m1, n1, s1, bi1, be1, out_f32, out_bf16, out_nchw, do_gelu);
}

// ---------------------------------------------------------------------------
// host launcher
// ---------------------------------------------------------------------------
extern "C" void kernel_launch(void* const* d_in, const int* in_sizes, int n_in,
                              void* d_out, int out_size, void* d_ws, size_t ws_size,
                              hipStream_t stream) {
  const float* x      = (const float*)d_in[0];
  const float* cpe_w  = (const float*)d_in[1];
  const float* cpe_b  = (const float*)d_in[2];
  const float* fc1_w  = (const float*)d_in[3];
  const float* fc1_b  = (const float*)d_in[4];
  const float* fc1_g  = (const float*)d_in[5];
  const float* fc1_be = (const float*)d_in[6];
  const float* loc_w  = (const float*)d_in[7];
  const float* loc_b  = (const float*)d_in[8];
  const float* loc_g  = (const float*)d_in[9];
  const float* loc_be = (const float*)d_in[10];
  const float* na_w   = (const float*)d_in[11];
  const float* na_b   = (const float*)d_in[12];
  const float* na_g   = (const float*)d_in[13];
  const float* na_be  = (const float*)d_in[14];
  const float* gc_w   = (const float*)d_in[15];
  const float* gc_b   = (const float*)d_in[16];
  const float* gc_g   = (const float*)d_in[17];
  const float* gc_be  = (const float*)d_in[18];
  const float* fc2_w  = (const float*)d_in[19];
  const float* fc2_b  = (const float*)d_in[20];
  const float* fc2_g  = (const float*)d_in[21];
  const float* fc2_be = (const float*)d_in[22];

  char* ws = (char*)d_ws;
  size_t off = 0;
  auto carve = [&](size_t bytes) -> void* {
    off = (off + 255) & ~(size_t)255;
    void* p = ws + off;
    off += bytes;
    return p;
  };
  float* xn    = (float*)carve((size_t)PTOT * CC * 4);   // NHWC input
  float* xact  = (float*)carve((size_t)PTOT * CC * 4);   // fp32 activations (reused)
  bf16*  A1    = (bf16*) carve((size_t)PTOT * CC * 2);   // fc1 operand
  bf16*  Abig  = (bf16*) carve((size_t)PTOT * 512 * 2);  // concat operands (reused)
  bf16*  A5    = (bf16*) carve((size_t)PTOT * CC * 2);   // fc2 operand
  bf16*  wbf   = (bf16*) carve((size_t)524288 * 2);      // all weights, bf16
  float* thrb  = (float*)carve(256);

  bf16* w_fc1 = wbf;
  bf16* w_loc = wbf + 65536;
  bf16* w_na  = wbf + 65536 + 131072;
  bf16* w_gc  = wbf + 65536 + 2 * 131072;
  bf16* w_fc2 = wbf + 65536 + 3 * 131072;

  dim3 gB(256, 1, 1);
  dim3 gGemm(4, PTOT / 128, 1);   // N tiles x M tiles

  k_transpose<<<PTOT, gB, 0, stream>>>(x, xn);
  k_f32_to_bf16<<<(65536 + 255) / 256, gB, 0, stream>>>(fc1_w, w_fc1, 65536);
  k_f32_to_bf16<<<(131072 + 255) / 256, gB, 0, stream>>>(loc_w, w_loc, 131072);
  k_f32_to_bf16<<<(131072 + 255) / 256, gB, 0, stream>>>(na_w, w_na, 131072);
  k_f32_to_bf16<<<(131072 + 255) / 256, gB, 0, stream>>>(gc_w, w_gc, 131072);
  k_f32_to_bf16<<<(65536 + 255) / 256, gB, 0, stream>>>(fc2_w, w_fc2, 65536);

  // CPE + residual -> A1
  k_cpe<<<PTOT, gB, 0, stream>>>(xn, cpe_w, cpe_b, A1);
  // fc1 + BN -> xact
  k_gemm<<<gGemm, gB, 0, stream>>>(A1, w_fc1, fc1_b, fc1_g, fc1_be,
                                   xact, nullptr, nullptr, 256, 0);
  // local MR aggregate -> Abig
  k_thr<<<BB, gB, 0, stream>>>(xact, thrb);
  k_mr<<<PTOT, gB, 0, stream>>>(xact, thrb, Abig, 0);
  // loc conv + BN + GELU -> xact
  k_gemm<<<gGemm, gB, 0, stream>>>(Abig, w_loc, loc_b, loc_g, loc_be,
                                   xact, nullptr, nullptr, 512, 1);
  // NodeAttention -> Abig
  k_na<<<PTOT, gB, 0, stream>>>(xact, Abig);
  // na conv + BN + GELU -> xact
  k_gemm<<<gGemm, gB, 0, stream>>>(Abig, w_na, na_b, na_g, na_be,
                                   xact, nullptr, nullptr, 512, 1);
  // global MR aggregate -> Abig
  k_thr<<<BB, gB, 0, stream>>>(xact, thrb);
  k_mr<<<PTOT, gB, 0, stream>>>(xact, thrb, Abig, 1);
  // gc conv + BN + GELU -> A5 (bf16 only)
  k_gemm<<<gGemm, gB, 0, stream>>>(Abig, w_gc, gc_b, gc_g, gc_be,
                                   nullptr, A5, nullptr, 512, 1);
  // fc2 + BN -> d_out (NCHW fp32)
  k_gemm<<<gGemm, gB, 0, stream>>>(A5, w_fc2, fc2_b, fc2_g, fc2_be,
                                   nullptr, nullptr, (float*)d_out, 256, 0);
}
